// BiLSTM_73040213836432
// MI455X (gfx1250) — matrix operand order repaired and checked by reading, not compiled, not run on hardware
//
#include <hip/hip_runtime.h>
#include <hip/hip_bf16.h>
#include <stdint.h>

// ---------------------------------------------------------------------------
// MI455X (gfx1250) BiLSTM: all GEMMs via v_wmma_f32_16x16x32_bf16 (wave32).
//  - input/output GEMMs: parallel WMMA tiles, LN fused via LDS + shfl reduce
//  - recurrent scan: 2 persistent workgroups (one per direction), h/c state
//    resident in LDS (320KB/WGP), Whh streamed from L2 (192MB -> resident)
//  - all half-selection conditions are wave-uniform / unroll-constant so no
//    EXEC-mask divergence surrounds the WMMA pipeline
// ---------------------------------------------------------------------------

typedef __bf16 v16bf __attribute__((ext_vector_type(16)));
typedef __bf16 v8bf  __attribute__((ext_vector_type(8)));
typedef float  v8f   __attribute__((ext_vector_type(8)));
typedef float  v4f   __attribute__((ext_vector_type(4)));

#define LN_EPS 1e-5f

static_assert(sizeof(__bf16) == 2, "bf16 size");

__device__ __forceinline__ __bf16 f2bf(float f) {
  unsigned u = __builtin_bit_cast(unsigned int, f);
  u += 0x7FFFu + ((u >> 16) & 1u);          // round-to-nearest-even
  unsigned short h = (unsigned short)(u >> 16);
  return __builtin_bit_cast(__bf16, h);
}
__device__ __forceinline__ float bf2f(__bf16 b) {
  unsigned short h = __builtin_bit_cast(unsigned short, b);
  unsigned u = ((unsigned)h) << 16;
  return __builtin_bit_cast(float, u);
}

__device__ __forceinline__ v8f wmma_bf16(v16bf a, v16bf b, v8f c) {
  // D = A(16x32 bf16) * B(32x16 bf16) + C(16x16 f32)
  return __builtin_amdgcn_wmma_f32_16x16x32_bf16(
      /*neg_a=*/false, a, /*neg_b=*/false, b,
      /*c_mod=*/(short)0, c, /*reuse_a=*/false, /*reuse_b=*/false);
}

// A fragment (16-bit, 16x32): lane m=lane&15 holds
// K = kbase..kbase+15, kbase = (lane&16)?16:0 (matches contiguous B).
// `p` points at element (m, kstep*32 + ((lane&16)?16:0)) of a row-major
// [M][ldK] bf16 array (16B aligned -> two b128 loads).
__device__ __forceinline__ v16bf load_a_frag(const __bf16* p) {
  v8bf lo = *(const v8bf*)(p);
  v8bf hi = *(const v8bf*)(p + 8);
  return __builtin_shufflevector(lo, hi, 0, 1, 2, 3, 4, 5, 6, 7,
                                 8, 9, 10, 11, 12, 13, 14, 15);
}

// full-wave (32) and half-wave (16) butterfly reductions
__device__ __forceinline__ float wred_sum32(float v) {
  #pragma unroll
  for (int m = 16; m >= 1; m >>= 1) v += __shfl_xor(v, m, 32);
  return v;
}
__device__ __forceinline__ float hred_sum16(float v) {
  #pragma unroll
  for (int m = 8; m >= 1; m >>= 1) v += __shfl_xor(v, m, 32);
  return v;
}
__device__ __forceinline__ float hred_max16(float v) {
  #pragma unroll
  for (int m = 8; m >= 1; m >>= 1) v = fmaxf(v, __shfl_xor(v, m, 32));
  return v;
}

// ---------------------------------------------------------------------------
// K0: f32 -> bf16 conversion (weights into workspace). n % 4 == 0.
// ---------------------------------------------------------------------------
__global__ void k_f32_to_bf16(const float* __restrict__ src,
                              __bf16* __restrict__ dst, int n4) {
  int i = blockIdx.x * blockDim.x + threadIdx.x;
  if (i < n4) {
    v4f x = *(const v4f*)(src + (size_t)i * 4);
    __bf16* d = dst + (size_t)i * 4;
    #pragma unroll
    for (int j = 0; j < 4; ++j) d[j] = f2bf(x[j]);
  }
}

// ---------------------------------------------------------------------------
// K1: embedding gather + dual input GEMM + fused LN.
// grid = 1024 blocks (one 16-row M tile each), 512 threads = 16 waves.
// Each block computes full 2048 cols (ig_e | ig_r) so LN is block-local.
// Waves 0..7 -> ig_e cols, waves 8..15 -> ig_r cols (wave-uniform select).
// dyn LDS: A tile 16x256 bf16 (8KB) + gate buffer 16x2048 f32 (128KB)
// ---------------------------------------------------------------------------
__global__ void __launch_bounds__(512)
k_embed_gemm_ln(const int* __restrict__ xidx, const float* __restrict__ emb,
                const __bf16* __restrict__ WihE, const __bf16* __restrict__ WihR,
                const float* __restrict__ bihE, const float* __restrict__ bihR,
                __bf16* __restrict__ igE, __bf16* __restrict__ igR) {
  extern __shared__ char smem[];
  __bf16* alds = (__bf16*)smem;                  // [16][256]
  float*  gbuf = (float*)(smem + 16 * 256 * 2);  // [16][2048]

  const int tid = threadIdx.x;
  const int lane = tid & 31;
  const int w = tid >> 5;          // wave 0..15
  const int row0 = blockIdx.x * 16;
  const int ln = lane & 15;
  const int hio = (lane & 16) ? 1 : 0;

  // gather 16 embedding rows -> bf16 LDS A tile (8 elems/thread, vectorized)
  {
    int r = tid >> 5;                 // 0..15
    int c0 = (tid & 31) * 8;
    const float* e = emb + (size_t)xidx[row0 + r] * 256 + c0;
    v4f e0 = *(const v4f*)(e);
    v4f e1 = *(const v4f*)(e + 4);
    v8bf o;
    #pragma unroll
    for (int j = 0; j < 4; ++j) { o[j] = f2bf(e0[j]); o[4 + j] = f2bf(e1[j]); }
    *(v8bf*)(alds + r * 256 + c0) = o;
  }
  __syncthreads();

  // wave-uniform half selection: whole wave's 128-col range is in one half
  const __bf16* Wsel = (w < 8) ? WihE : WihR;
  const float*  bsel = (w < 8) ? bihE : bihR;
  const int ch = (w & 7) * 128;     // col within selected half

  // hoist per-lane B row pointers (row = ch + j*16 + ln, K offset hio*16)
  const __bf16* wp[8];
  #pragma unroll
  for (int j = 0; j < 8; ++j)
    wp[j] = Wsel + (size_t)(ch + j * 16 + ln) * 256 + hio * 16;
  const __bf16* aB = alds + ln * 256 + hio * 16;

  v8f acc[8] = {};
  #pragma unroll
  for (int kk = 0; kk < 8; ++kk) {
    v16bf afr = load_a_frag(aB + kk * 32);
    #pragma unroll
    for (int j = 0; j < 8; ++j) {
      v16bf bfr = *(const v16bf*)(wp[j] + kk * 32);
      acc[j] = wmma_bf16(afr, bfr, acc[j]);
    }
  }
  // scatter accumulators (+bias) into LDS gate buffer
  float* gp = gbuf + (size_t)(hio * 8) * 2048 + w * 128 + ln;
  #pragma unroll
  for (int j = 0; j < 8; ++j) {
    float bias = bsel[ch + j * 16 + ln];
    #pragma unroll
    for (int r = 0; r < 8; ++r)
      gp[r * 2048 + j * 16] = acc[j][r] + bias;
  }
  __syncthreads();

  // fused LN over rows of 1024: 32 tasks (16 rows x {e,r}), 2 per wave
  #pragma unroll
  for (int t = 0; t < 2; ++t) {
    const float* src = gbuf + w * 2048 + t * 1024 + lane * 32;
    v4f v[8];
    float s = 0.f, s2 = 0.f;
    #pragma unroll
    for (int q = 0; q < 8; ++q) {
      v[q] = *(const v4f*)(src + q * 4);
      #pragma unroll
      for (int e = 0; e < 4; ++e) { s += v[q][e]; s2 += v[q][e] * v[q][e]; }
    }
    s = wred_sum32(s); s2 = wred_sum32(s2);
    float mean = s * (1.0f / 1024.0f);
    float rs = rsqrtf(s2 * (1.0f / 1024.0f) - mean * mean + LN_EPS);
    __bf16* dst = (t ? igR : igE) + (size_t)(row0 + w) * 1024 + lane * 32;
    #pragma unroll
    for (int q = 0; q < 4; ++q) {
      v8bf o;
      #pragma unroll
      for (int e = 0; e < 4; ++e) {
        o[e]     = f2bf((v[2 * q][e]     - mean) * rs);
        o[4 + e] = f2bf((v[2 * q + 1][e] - mean) * rs);
      }
      *(v8bf*)(dst + q * 8) = o;
    }
  }
}

// ---------------------------------------------------------------------------
// K3: LN-LSTM scan. grid=2 (dir), 1024 threads = 32 waves, persistent.
// Per step: 32x1024 WMMA GEMM (K=256) from LDS-resident bf16 h state,
// then per-batch-row LN + gates + cell LN (wave w owns batch row w).
// All B pointers / biases hoisted out of the 512-iteration loop; inner
// loop is pure immediate-offset loads + WMMA.
// dyn LDS: gbuf 32x1024 f32 (128KB) + hbuf 32x256 bf16 (16KB)
//          + cbuf 32x256 f32 (32KB)
// ---------------------------------------------------------------------------
__global__ void __launch_bounds__(1024)
k_lstm_scan(const __bf16* __restrict__ igE, const __bf16* __restrict__ igR,
            const __bf16* __restrict__ WhhE, const __bf16* __restrict__ WhhR,
            const float* __restrict__ bhhE, const float* __restrict__ bhhR,
            __bf16* __restrict__ hsE, __bf16* __restrict__ hsR) {
  extern __shared__ char smem[];
  float*  gbuf = (float*)smem;                                  // [32][1024]
  __bf16* hbuf = (__bf16*)(smem + 32 * 1024 * 4);               // [32][256]
  float*  cbuf = (float*)(smem + 32 * 1024 * 4 + 32 * 256 * 2); // [32][256]

  const int dir = blockIdx.x;
  const __bf16* ig  = dir ? igR  : igE;
  const __bf16* Whh = dir ? WhhR : WhhE;
  const float*  bhh = dir ? bhhR : bhhE;
  __bf16* hs = dir ? hsR : hsE;

  const int tid = threadIdx.x, lane = tid & 31, w = tid >> 5;
  const int ln = lane & 15, hio = (lane & 16) ? 1 : 0;

  // h0 = c0 = 0
  for (int i = tid; i < 32 * 256; i += 1024) {
    hbuf[i] = f2bf(0.f);
    cbuf[i] = 0.f;
  }
  __syncthreads();

  const int m0 = (w & 1) * 16;        // M tile (batch rows)
  const int ntb = (w >> 1) * 4;       // 4 n-tiles per wave (64 total)

  // ---- loop-invariant pointers / biases -------------------------------
  const __bf16* wp[4];
  float biasj[4];
  #pragma unroll
  for (int j = 0; j < 4; ++j) {
    int c = (ntb + j) * 16 + ln;
    wp[j] = Whh + (size_t)c * 256 + hio * 16;
    biasj[j] = bhh[c];
  }
  const __bf16* aB = hbuf + (m0 + ln) * 256 + hio * 16;
  float* gp = gbuf + (size_t)(m0 + hio * 8) * 1024 + ntb * 16 + ln;
  const float* gln = gbuf + w * 1024 + lane * 32;   // LN stats source (row w)
  const float* gd  = gbuf + w * 1024 + lane * 8;    // gate source (row w)
  float*  cb  = cbuf + w * 256 + lane * 8;
  __bf16* hbl = hbuf + w * 256 + lane * 8;
  const __bf16* igbase = ig + (size_t)w * 512 * 1024;

  for (int t = 0; t < 512; ++t) {
    const int te = dir ? (511 - t) : t;
    const __bf16* igrow = igbase + (size_t)te * 1024;
    // warm the 2KB gate row while the GEMM runs (global_prefetch_b8)
    __builtin_prefetch(igrow + lane * 32, 0, 0);

    // hg = h @ Whh^T : 2 M-tiles x 64 N-tiles x 8 k-steps = 1024 WMMA/step
    v8f acc[4] = {};
    #pragma unroll
    for (int kk = 0; kk < 8; ++kk) {
      v16bf afr = load_a_frag(aB + kk * 32);
      #pragma unroll
      for (int j = 0; j < 4; ++j) {
        v16bf bfr = *(const v16bf*)(wp[j] + kk * 32);
        acc[j] = wmma_bf16(afr, bfr, acc[j]);
      }
    }
    #pragma unroll
    for (int j = 0; j < 4; ++j) {
      #pragma unroll
      for (int r = 0; r < 8; ++r)
        gp[r * 1024 + j * 16] = acc[j][r] + biasj[j];
    }
    __syncthreads();

    // wave w handles batch row b=w: LN(hg), gates, cell LN, h update
    {
      float s = 0.f, s2 = 0.f;
      #pragma unroll
      for (int q = 0; q < 8; ++q) {
        v4f x = *(const v4f*)(gln + q * 4);
        #pragma unroll
        for (int e = 0; e < 4; ++e) { s += x[e]; s2 += x[e] * x[e]; }
      }
      s = wred_sum32(s); s2 = wred_sum32(s2);
      float mean = s * (1.f / 1024.f);
      float rs = rsqrtf(s2 * (1.f / 1024.f) - mean * mean + LN_EPS);

      // vector loads: hg gate groups (LDS) + precomputed input gates (global)
      float hgv[4][8];
      #pragma unroll
      for (int gidx = 0; gidx < 4; ++gidx) {
        v4f a0 = *(const v4f*)(gd + gidx * 256);
        v4f a1 = *(const v4f*)(gd + gidx * 256 + 4);
        #pragma unroll
        for (int e = 0; e < 4; ++e) { hgv[gidx][e] = a0[e]; hgv[gidx][4 + e] = a1[e]; }
      }
      v8bf ig_i = *(const v8bf*)(igrow + lane * 8);
      v8bf ig_f = *(const v8bf*)(igrow + 256 + lane * 8);
      v8bf ig_g = *(const v8bf*)(igrow + 512 + lane * 8);
      v8bf ig_o = *(const v8bf*)(igrow + 768 + lane * 8);
      v4f cold0 = *(const v4f*)(cb);
      v4f cold1 = *(const v4f*)(cb + 4);

      float av[8], ov[8];
      float cs = 0.f, cs2 = 0.f;
      #pragma unroll
      for (int j = 0; j < 8; ++j) {
        float gi = (hgv[0][j] - mean) * rs + bf2f(ig_i[j]);
        float gf = (hgv[1][j] - mean) * rs + bf2f(ig_f[j]);
        float gg = (hgv[2][j] - mean) * rs + bf2f(ig_g[j]);
        float go = (hgv[3][j] - mean) * rs + bf2f(ig_o[j]);
        float iv = 1.f / (1.f + __expf(-gi));
        float fv = 1.f / (1.f + __expf(-gf));
        float gv = tanhf(gg);
        ov[j] = 1.f / (1.f + __expf(-go));
        float cold = (j < 4) ? cold0[j & 3] : cold1[j & 3];
        float a = fv * cold + iv * gv;
        av[j] = a; cs += a; cs2 += a * a;
      }
      cs = wred_sum32(cs); cs2 = wred_sum32(cs2);
      float cm = cs * (1.f / 256.f);
      float crs = rsqrtf(cs2 * (1.f / 256.f) - cm * cm + LN_EPS);

      v4f cw0, cw1;
      v8bf hw;
      #pragma unroll
      for (int j = 0; j < 8; ++j) {
        float cy = (av[j] - cm) * crs;
        if (j < 4) cw0[j & 3] = cy; else cw1[j & 3] = cy;
        hw[j] = f2bf(ov[j] * tanhf(cy));
      }
      *(v4f*)(cb) = cw0;
      *(v4f*)(cb + 4) = cw1;
      *(v8bf*)(hbl) = hw;                                   // next-step A
      *(v8bf*)(hs + ((size_t)te * 32 + w) * 256 + lane * 8) = hw;
    }
    __syncthreads();
  }
}

// ---------------------------------------------------------------------------
// K4: logits = [xe|xr] @ Wout^T + bout, then log_softmax over T=48.
// grid = 128 blocks x 256 threads (8 waves); wave = one 16-row M tile,
// 3 N-tiles (48 cols), K = 512 (16 k-steps). kk<8 selects hs_e vs hs_r at
// unroll time (no divergence). Softmax via 16-lane shfl reductions.
// ---------------------------------------------------------------------------
__global__ void __launch_bounds__(256)
k_out_logsoftmax(const __bf16* __restrict__ hsE, const __bf16* __restrict__ hsR,
                 const __bf16* __restrict__ Wout, const float* __restrict__ bout,
                 float* __restrict__ out) {
  const int tid = threadIdx.x, lane = tid & 31, w = tid >> 5;
  const int ln = lane & 15, hio = (lane & 16) ? 1 : 0;
  const int m0 = blockIdx.x * 128 + w * 16;

  // this lane's A row: m = b*512 + l ; h stored as [l][b][d]
  const int m = m0 + ln;
  const int b = m >> 9;
  const int l = m & 511;
  const __bf16* ha = hsE + ((size_t)l * 32 + b) * 256 + hio * 16;
  const __bf16* hb = hsR + ((size_t)l * 32 + b) * 256 + hio * 16;

  const __bf16* wp[3];
  #pragma unroll
  for (int j = 0; j < 3; ++j)
    wp[j] = Wout + (size_t)(j * 16 + ln) * 512 + hio * 16;

  v8f acc[3] = {};
  #pragma unroll
  for (int kk = 0; kk < 16; ++kk) {
    const __bf16* p = (kk < 8) ? (ha + kk * 32) : (hb + (kk - 8) * 32);
    v16bf afr = load_a_frag(p);
    #pragma unroll
    for (int j = 0; j < 3; ++j) {
      v16bf bfr = *(const v16bf*)(wp[j] + kk * 32);
      acc[j] = wmma_bf16(afr, bfr, acc[j]);
    }
  }

  const float b0 = bout[ln], b1 = bout[16 + ln], b2 = bout[32 + ln];
  #pragma unroll
  for (int r = 0; r < 8; ++r) {
    float v0 = acc[0][r] + b0;
    float v1 = acc[1][r] + b1;
    float v2 = acc[2][r] + b2;
    float mx = hred_max16(fmaxf(v0, fmaxf(v1, v2)));
    float se = hred_sum16(__expf(v0 - mx) + __expf(v1 - mx) + __expf(v2 - mx));
    float lse = mx + __logf(se);
    float* o = out + (size_t)(m0 + hio * 8 + r) * 48;
    o[ln] = v0 - lse;
    o[16 + ln] = v1 - lse;
    o[32 + ln] = v2 - lse;
  }
}

// ---------------------------------------------------------------------------
extern "C" void kernel_launch(void* const* d_in, const int* in_sizes, int n_in,
                              void* d_out, int out_size, void* d_ws,
                              size_t ws_size, hipStream_t stream) {
  (void)in_sizes; (void)n_in; (void)out_size; (void)ws_size;
  const int*   x     = (const int*)d_in[0];
  const float* emb   = (const float*)d_in[1];
  const float* WihE  = (const float*)d_in[2];
  const float* bihE  = (const float*)d_in[3];
  const float* WhhE  = (const float*)d_in[4];
  const float* bhhE  = (const float*)d_in[5];
  const float* WihR  = (const float*)d_in[6];
  const float* bihR  = (const float*)d_in[7];
  const float* WhhR  = (const float*)d_in[8];
  const float* bhhR  = (const float*)d_in[9];
  const float* Wout  = (const float*)d_in[10];
  const float* bout  = (const float*)d_in[11];
  float* out = (float*)d_out;

  char* ws = (char*)d_ws;
  size_t off = 0;
  auto alloc = [&](size_t bytes) -> void* {
    void* p = ws + off;
    off = (off + bytes + 255) & ~(size_t)255;
    return p;
  };
  __bf16* WihE_b = (__bf16*)alloc(1024 * 256 * 2);
  __bf16* WhhE_b = (__bf16*)alloc(1024 * 256 * 2);
  __bf16* WihR_b = (__bf16*)alloc(1024 * 256 * 2);
  __bf16* WhhR_b = (__bf16*)alloc(1024 * 256 * 2);
  __bf16* Wout_b = (__bf16*)alloc(48 * 512 * 2);
  __bf16* igE    = (__bf16*)alloc((size_t)16384 * 1024 * 2);  // 32MB
  __bf16* igR    = (__bf16*)alloc((size_t)16384 * 1024 * 2);  // 32MB
  __bf16* hsE    = (__bf16*)alloc((size_t)512 * 32 * 256 * 2);
  __bf16* hsR    = (__bf16*)alloc((size_t)512 * 32 * 256 * 2);

  const int nW4 = (1024 * 256) / 4;
  k_f32_to_bf16<<<(nW4 + 255) / 256, 256, 0, stream>>>(WihE, WihE_b, nW4);
  k_f32_to_bf16<<<(nW4 + 255) / 256, 256, 0, stream>>>(WhhE, WhhE_b, nW4);
  k_f32_to_bf16<<<(nW4 + 255) / 256, 256, 0, stream>>>(WihR, WihR_b, nW4);
  k_f32_to_bf16<<<(nW4 + 255) / 256, 256, 0, stream>>>(WhhR, WhhR_b, nW4);
  const int nO4 = (48 * 512) / 4;
  k_f32_to_bf16<<<(nO4 + 255) / 256, 256, 0, stream>>>(Wout, Wout_b, nO4);

  size_t smem1 = 16 * 256 * 2 + 16 * 2048 * 4;            // 136KB
  k_embed_gemm_ln<<<1024, 512, smem1, stream>>>(x, emb, WihE_b, WihR_b, bihE,
                                                bihR, igE, igR);

  size_t smem3 = 32 * 1024 * 4 + 32 * 256 * 2 + 32 * 256 * 4;  // 176KB
  k_lstm_scan<<<2, 1024, smem3, stream>>>(igE, igR, WhhE_b, WhhR_b, bhhE, bhhR,
                                          hsE, hsR);

  k_out_logsoftmax<<<128, 256, 0, stream>>>(hsE, hsR, Wout_b, bout, out);
}
